// MultiHeadedAttention_79491254714609
// MI455X (gfx1250) — matrix operand order repaired and checked by reading, not compiled, not run on hardware
//
#include <hip/hip_runtime.h>

// ---------------------------------------------------------------------------
// MultiHeadedAttention for MI455X (gfx1250, wave32, WMMA bf16 / f32 accum)
// Round 4: TDM (tensor_load_to_lds, 6-arg form) staging for GEMM A-tiles.
// ---------------------------------------------------------------------------

typedef __attribute__((ext_vector_type(16))) __bf16        v16bf;
typedef __attribute__((ext_vector_type(8)))  float         v8f;
typedef __attribute__((ext_vector_type(8)))  unsigned int  v8u;
typedef __attribute__((ext_vector_type(4)))  unsigned int  v4ui;
typedef __attribute__((ext_vector_type(8)))  int           v8i;
typedef __attribute__((ext_vector_type(4)))  int           v4i;

#define DEV static __device__ __forceinline__

constexpr int    kB    = 4;
constexpr int    kS    = 1024;
constexpr int    kD    = 1024;
constexpr int    kH    = 16;
constexpr int    kDk   = 64;
constexpr int    kRows = kB * kS;            // 4096
constexpr size_t kSSq  = (size_t)kS * kS;    // 1M

#if defined(__HIP_DEVICE_COMPILE__) && \
    __has_builtin(__builtin_amdgcn_tensor_load_to_lds) && \
    __has_builtin(__builtin_amdgcn_s_wait_tensorcnt)
#define HAVE_TDM 1
#else
#define HAVE_TDM 0
#endif

// float -> bf16 bits, round-to-nearest-even
DEV unsigned short f2bf(float f) {
  unsigned u = __builtin_bit_cast(unsigned, f);
  unsigned r = 0x7FFFu + ((u >> 16) & 1u);
  return (unsigned short)((u + r) >> 16);
}

// A-fragment (16x32 bf16, MxK): lane<16 -> K base 0, lane>=16 -> +8;
// regs 0..3 cover K 0..7(+8), regs 4..7 cover K 16..23(+8). Pairs contiguous.
DEV v16bf frag_a(const unsigned short* base, int stride, int lane) {
  const int m  = lane & 15;
  const int kh = (lane < 16) ? 0 : 8;
  const unsigned short* rp = base + (size_t)m * stride;
  v8u r;
#pragma unroll
  for (int i = 0; i < 8; ++i) {
    int k = ((i < 4) ? 0 : 16) + kh + (i & 3) * 2;
    r[i] = *(const unsigned int*)(rp + k);
  }
  return __builtin_bit_cast(v16bf, r);
}

// B-fragment (32x16 bf16, KxN): same K order as A (lane>=16 -> +8; regs 4..7 -> K 16..),
// source is row-major [N][K] so pairs are contiguous.
DEV v16bf frag_b(const unsigned short* base, int stride, int lane) {
  const int n  = lane & 15;
  const int kh = (lane < 16) ? 0 : 8;
  const unsigned short* rp = base + (size_t)n * stride;
  v8u r;
#pragma unroll
  for (int i = 0; i < 8; ++i) {
    r[i] = *(const unsigned int*)(rp + ((i < 4) ? 0 : 16) + kh + (i & 3) * 2);
  }
  return __builtin_bit_cast(v16bf, r);
}

DEV v8f wmma_bf16(v16bf a, v16bf b, v8f c) {
  return __builtin_amdgcn_wmma_f32_16x16x32_bf16(
      /*neg_a=*/false, a, /*neg_b=*/false, b,
      /*c_mod=*/(short)0, c, /*reuse_a=*/false, /*reuse_b=*/false);
}

struct __align__(16) U128 { unsigned int u[4]; };
union  U128S { U128 v; unsigned short s[8]; };

// ---------------------------------------------------------------------------
// fp32 -> bf16-bits cast
// ---------------------------------------------------------------------------
__global__ __launch_bounds__(256) void k_cast(const float* __restrict__ in,
                                              unsigned short* __restrict__ out,
                                              int n) {
  int i = blockIdx.x * 256 + threadIdx.x;
  if (i < n) out[i] = f2bf(in[i]);
}

// ---------------------------------------------------------------------------
// Dense GEMM: Y = X[M,K] * W[K,N] + bias, bf16 WMMA, f32 accum.
//   mode 0: bf16 out, head-split     [b][h][s][dk]
//   mode 1: bf16 out, head-split+T   [b][h][dk][s]   (for V)
//   mode 2: f32  out, row-major      [m][n]          (final projection)
// Block: 256 thr (8 waves), 128x128 tile; wave = 32x64 (2x4 WMMA subtiles).
// A tile staged with a single TDM DMA per K-step (hardware LDS padding
// recreates the 40-element bank-conflict-free row stride); B tile goes
// through VGPRs because it is transposed on the LDS store.
// ---------------------------------------------------------------------------
__global__ __launch_bounds__(256) void k_gemm(
    const unsigned short* __restrict__ X, const unsigned short* __restrict__ W,
    const float* __restrict__ bias, unsigned short* __restrict__ ybf,
    float* __restrict__ yf, int M, int N, int K, int mode) {
  __shared__ unsigned short lds_a[128][40];   // [m][k], 64B row + 16B pad
  __shared__ unsigned short lds_bt[128][40];  // [n][k] (transposed on store)

  const int tid  = threadIdx.x;
  const int lane = tid & 31;
  const int wid  = tid >> 5;
  const int wm   = wid & 3;    // 32-row strip
  const int wn   = wid >> 2;   // 64-col strip
  const int m0   = blockIdx.y * 128;
  const int n0   = blockIdx.x * 128;

  v8f acc[2][4];
#pragma unroll
  for (int i = 0; i < 2; ++i)
#pragma unroll
    for (int j = 0; j < 4; ++j) acc[i][j] = (v8f){};

  for (int k0 = 0; k0 < K; k0 += 32) {
    if (k0) __syncthreads();

#if HAVE_TDM
    if (wid == 0) {
      // ---- Tensor DMA descriptor (D#) for the 128x32 bf16 A tile ----
      unsigned long long ga =
          (unsigned long long)(const void*)(X + (size_t)m0 * K + k0);
      unsigned lds_off = (unsigned)(unsigned long long)(const void*)&lds_a[0][0];
      v4ui g0;
      g0[0] = 1u;                                   // count=1, user descriptor
      g0[1] = lds_off;                              // lds_addr (bytes)
      g0[2] = (unsigned)(ga & 0xFFFFFFFFu);         // global_addr[31:0]
      g0[3] = (unsigned)((ga >> 32) & 0x1FFFFFFu)   // global_addr[56:32]
              | (2u << 30);                         // type=2 ("image")
      v8i g1;
      g1[0] = (int)((1u << 16)      // data_size = 2 bytes
                    | (1u << 20)    // pad_enable
                    | (3u << 22)    // pad_interval: every 16 dwords (64B row)
                    | (3u << 25));  // pad_amount: 4 dwords (16B) -> stride 80B
      g1[1] = (int)(((unsigned)K & 0xFFFFu) << 16);               // tensor_dim0 lo16
      g1[2] = (int)(((unsigned)K >> 16) |
                    (((unsigned)M & 0xFFFFu) << 16));             // dim0 hi / dim1 lo
      g1[3] = (int)(((unsigned)M >> 16) | (32u << 16));           // dim1 hi / tile_dim0=32
      g1[4] = (int)128u;                                          // tile_dim1=128, tile_dim2=0
      g1[5] = (int)(unsigned)K;                                   // tensor_dim0_stride lo32
      g1[6] = 0;                                                  // stride hi / dim1_stride lo
      g1[7] = 0;
      v4i zero4 = {};
      v8i zero8 = {};
      __builtin_amdgcn_tensor_load_to_lds(g0, g1, zero4, zero4, zero8, 0);
      __builtin_amdgcn_s_wait_tensorcnt(0);
    }
#endif

#pragma unroll
    for (int t = 0; t < 2; ++t) {
      int c = tid + t * 256;                      // 0..511 chunk id
#if !HAVE_TDM
      int ar = c >> 2, akq = (c & 3) * 8;         // A: 128 rows x 4 chunks
      U128 av = *(const U128*)(X + (size_t)(m0 + ar) * K + k0 + akq);
      *(U128*)(&lds_a[ar][akq]) = av;             // 16B aligned
#endif
      int br = c >> 4, bnq = (c & 15) * 8;        // B: 32 rows x 16 chunks
      U128S bu;
      bu.v = *(const U128*)(W + (size_t)(k0 + br) * N + n0 + bnq);
#pragma unroll
      for (int j = 0; j < 8; ++j) lds_bt[bnq + j][br] = bu.s[j];
    }
    if (k0 + 32 < K)
      __builtin_prefetch(X + (size_t)(m0 + (tid >> 2)) * K + k0 + 32, 0, 1);
    __syncthreads();

    v16bf a[2], b[4];
#pragma unroll
    for (int i = 0; i < 2; ++i) a[i] = frag_a(&lds_a[wm * 32 + i * 16][0], 40, lane);
#pragma unroll
    for (int j = 0; j < 4; ++j) b[j] = frag_b(&lds_bt[wn * 64 + j * 16][0], 40, lane);
#pragma unroll
    for (int i = 0; i < 2; ++i)
#pragma unroll
      for (int j = 0; j < 4; ++j) acc[i][j] = wmma_bf16(a[i], b[j], acc[i][j]);
  }

  const int moff = (lane < 16) ? 0 : 8;
  const int nl   = lane & 15;
#pragma unroll
  for (int i = 0; i < 2; ++i) {
#pragma unroll
    for (int j = 0; j < 4; ++j) {
      int n = n0 + wn * 64 + j * 16 + nl;
      float bvv = bias ? bias[n] : 0.f;
#pragma unroll
      for (int e = 0; e < 8; ++e) {
        int m = m0 + wm * 32 + i * 16 + moff + e;
        float v = acc[i][j][e] + bvv;
        if (mode == 2) {
          yf[(size_t)m * N + n] = v;
        } else {
          int bb = m >> 10, s = m & 1023;
          int hh = n >> 6,  d = n & 63;
          size_t addr = (mode == 0)
              ? ((((size_t)bb * kH + hh) * kS + s) * kDk + d)
              : ((((size_t)bb * kH + hh) * kDk + d) * kS + s);
          ybf[addr] = f2bf(v);
        }
      }
    }
  }
}

// ---------------------------------------------------------------------------
// scores = mask ? (Q Kᵀ)/8 : -1e9 , per head. K=64 -> 2 WMMA k-steps,
// fragments loaded directly from global (no LDS needed).
// Output fp32 into the scores region of d_out: [b][h][sq][sk].
// ---------------------------------------------------------------------------
__global__ __launch_bounds__(256) void k_scores(
    const unsigned short* __restrict__ Q,   // [b][h][s][dk] bf16
    const unsigned short* __restrict__ Kh,  // [b][h][s][dk] bf16
    const int* __restrict__ mask,           // [b][sq][sk]
    float* __restrict__ sc) {               // [b][h][sq][sk]
  const int tid  = threadIdx.x;
  const int lane = tid & 31;
  const int wid  = tid >> 5;
  const int wm   = wid & 3, wn = wid >> 2;
  const int n0   = blockIdx.x * 128;
  const int m0   = blockIdx.y * 128;
  const int bh   = blockIdx.z;              // b*H + h
  const unsigned short* q  = Q  + (size_t)bh * kS * kDk;
  const unsigned short* kk = Kh + (size_t)bh * kS * kDk;

  v8f acc[2][4];
#pragma unroll
  for (int i = 0; i < 2; ++i)
#pragma unroll
    for (int j = 0; j < 4; ++j) acc[i][j] = (v8f){};

#pragma unroll
  for (int ks = 0; ks < kDk; ks += 32) {
    v16bf a[2], b[4];
#pragma unroll
    for (int i = 0; i < 2; ++i)
      a[i] = frag_a(q + (size_t)(m0 + wm * 32 + i * 16) * kDk + ks, kDk, lane);
#pragma unroll
    for (int j = 0; j < 4; ++j)
      b[j] = frag_b(kk + (size_t)(n0 + wn * 64 + j * 16) * kDk + ks, kDk, lane);
#pragma unroll
    for (int i = 0; i < 2; ++i)
#pragma unroll
      for (int j = 0; j < 4; ++j) acc[i][j] = wmma_bf16(a[i], b[j], acc[i][j]);
  }

  const int bb   = bh >> 4;
  const int moff = (lane < 16) ? 0 : 8;
  const int nl   = lane & 15;
#pragma unroll
  for (int i = 0; i < 2; ++i)
#pragma unroll
    for (int j = 0; j < 4; ++j) {
      int n = n0 + wn * 64 + j * 16 + nl;
#pragma unroll
      for (int e = 0; e < 8; ++e) {
        int m = m0 + wm * 32 + i * 16 + moff + e;
        int mv = mask[((size_t)bb * kS + m) * kS + n];
        float v = mv ? acc[i][j][e] * 0.125f : -1e9f;
        sc[((size_t)bh * kS + m) * kS + n] = v;
      }
    }
}

// ---------------------------------------------------------------------------
// Fold the linear conv chain:  Wc = Cw3*Cw2*Cw1 [16x32], bc = Cw3*(Cw2*Cb1+Cb2)+Cb3
// ---------------------------------------------------------------------------
__global__ __launch_bounds__(512) void k_foldconv(
    const float* __restrict__ Cw1, const float* __restrict__ Cb1,
    const float* __restrict__ Cw2, const float* __restrict__ Cb2,
    const float* __restrict__ Cw3, const float* __restrict__ Cb3,
    float* __restrict__ Wc, float* __restrict__ bc) {
  __shared__ float T[16 * 32];
  const int t = threadIdx.x;             // 0..511
  const int o = t >> 5, i = t & 31;
  {
    float s = 0.f;
#pragma unroll
    for (int j = 0; j < 32; ++j) s = fmaf(Cw2[o * 32 + j], Cw1[j * 32 + i], s);
    T[o * 32 + i] = s;
  }
  __syncthreads();
  {
    float s = 0.f;
#pragma unroll
    for (int m = 0; m < 16; ++m) s = fmaf(Cw3[o * 16 + m], T[m * 32 + i], s);
    Wc[o * 32 + i] = s;
  }
  if (t < 16) {
    float s = Cb3[t];
    for (int m = 0; m < 16; ++m) {
      float u = Cb2[m];
      for (int j = 0; j < 32; ++j) u = fmaf(Cw2[m * 32 + j], Cb1[j], u);
      s = fmaf(Cw3[t * 16 + m], u, s);
    }
    bc[t] = s;
  }
}

// ---------------------------------------------------------------------------
// Apply folded conv in place on the scores region.
// One thread owns one (b, pixel): reads its 16 masked-score + 16 matx
// channels, writes 16 output channels back to the same slots.
// ---------------------------------------------------------------------------
__global__ __launch_bounds__(256) void k_conv(
    float* __restrict__ sc, const float* __restrict__ matx,
    const float* __restrict__ Wc, const float* __restrict__ bc) {
  __shared__ float w[16 * 32];
  __shared__ float bb[16];
  for (int i = threadIdx.x; i < 512; i += 256) w[i] = Wc[i];
  if (threadIdx.x < 16) bb[threadIdx.x] = bc[threadIdx.x];
  __syncthreads();

  const size_t total = (size_t)kB * kSSq;
  for (size_t idx = (size_t)blockIdx.x * 256 + threadIdx.x; idx < total;
       idx += (size_t)gridDim.x * 256) {
    size_t b   = idx >> 20;                 // / (1024*1024)
    size_t pix = idx & (kSSq - 1);
    size_t base = b * kH * kSSq + pix;
    float x[32];
#pragma unroll
    for (int h = 0; h < 16; ++h) x[h]      = sc[base + (size_t)h * kSSq];
#pragma unroll
    for (int h = 0; h < 16; ++h) x[16 + h] = matx[base + (size_t)h * kSSq];
    float o[16];
#pragma unroll
    for (int oo = 0; oo < 16; ++oo) {
      float a = bb[oo];
#pragma unroll
      for (int i = 0; i < 32; ++i) a = fmaf(w[oo * 32 + i], x[i], a);
      o[oo] = a;
    }
#pragma unroll
    for (int h = 0; h < 16; ++h) sc[base + (size_t)h * kSSq] = o[h];
  }
}

// ---------------------------------------------------------------------------
// Softmax over last dim (1024): one wave32 per row, bf16 output.
// ---------------------------------------------------------------------------
__global__ __launch_bounds__(256) void k_softmax(const float* __restrict__ sc,
                                                 unsigned short* __restrict__ p) {
  const int lane = threadIdx.x & 31;
  const int wid  = threadIdx.x >> 5;
  const size_t row = (size_t)blockIdx.x * 8 + wid;   // B*H*S rows
  const float* r = sc + row * kS;
  float v[32];
  float mx = -3.4e38f;
#pragma unroll
  for (int i = 0; i < 32; ++i) {
    v[i] = r[lane + i * 32];
    mx = fmaxf(mx, v[i]);
  }
#pragma unroll
  for (int off = 16; off > 0; off >>= 1) mx = fmaxf(mx, __shfl_xor(mx, off, 32));
  float sum = 0.f;
#pragma unroll
  for (int i = 0; i < 32; ++i) {
    v[i] = __expf(v[i] - mx);
    sum += v[i];
  }
#pragma unroll
  for (int off = 16; off > 0; off >>= 1) sum += __shfl_xor(sum, off, 32);
  const float inv = 1.f / sum;
#pragma unroll
  for (int i = 0; i < 32; ++i) p[row * kS + lane + i * 32] = f2bf(v[i] * inv);
}

// ---------------------------------------------------------------------------
// x = P·V per head: M=1024(sq), N=64(d), K=1024(sk). V is pre-transposed
// [b][h][dk][s] so B-fragments are contiguous. Output bf16 [b*S+sq][h*64+d].
// ---------------------------------------------------------------------------
__global__ __launch_bounds__(256) void k_attnv(
    const unsigned short* __restrict__ P,   // [b][h][sq][sk] bf16
    const unsigned short* __restrict__ Vt,  // [b][h][dk][s]  bf16
    unsigned short* __restrict__ Xo) {      // [b*S][D] bf16
  const int lane = threadIdx.x & 31;
  const int wid  = threadIdx.x >> 5;
  const int m0   = blockIdx.x * 128 + wid * 16;
  const int bh   = blockIdx.y;
  const unsigned short* pp = P  + (size_t)bh * kSSq;
  const unsigned short* vt = Vt + (size_t)bh * kDk * kS;

  v8f acc[4];
#pragma unroll
  for (int j = 0; j < 4; ++j) acc[j] = (v8f){};

  for (int ks = 0; ks < kS; ks += 32) {
    v16bf a = frag_a(pp + (size_t)m0 * kS + ks, kS, lane);
#pragma unroll
    for (int j = 0; j < 4; ++j) {
      v16bf b = frag_b(vt + (size_t)(j * 16) * kS + ks, kS, lane);
      acc[j] = wmma_bf16(a, b, acc[j]);
    }
  }

  const int bb = bh >> 4, hh = bh & 15;
  const int moff = (lane < 16) ? 0 : 8;
  const int nl   = lane & 15;
#pragma unroll
  for (int j = 0; j < 4; ++j)
#pragma unroll
    for (int e = 0; e < 8; ++e) {
      int sq = m0 + moff + e;
      int d  = j * 16 + nl;
      Xo[((size_t)bb * kS + sq) * kD + hh * kDk + d] = f2bf(acc[j][e]);
    }
}

// ---------------------------------------------------------------------------
extern "C" void kernel_launch(void* const* d_in, const int* in_sizes, int n_in,
                              void* d_out, int out_size, void* d_ws, size_t ws_size,
                              hipStream_t stream) {
  const float* query  = (const float*)d_in[0];
  const float* key_in = (const float*)d_in[1];
  const float* value  = (const float*)d_in[2];
  const int*   mask   = (const int*)d_in[3];
  const float* matx   = (const float*)d_in[4];
  const float* Wq = (const float*)d_in[5];   const float* bq = (const float*)d_in[6];
  const float* Wk = (const float*)d_in[7];   const float* bk = (const float*)d_in[8];
  const float* Wv = (const float*)d_in[9];   const float* bv = (const float*)d_in[10];
  const float* Wo = (const float*)d_in[11];  const float* bo = (const float*)d_in[12];
  const float* Cw1 = (const float*)d_in[13]; const float* Cb1 = (const float*)d_in[14];
  const float* Cw2 = (const float*)d_in[15]; const float* Cb2 = (const float*)d_in[16];
  const float* Cw3 = (const float*)d_in[17]; const float* Cb3 = (const float*)d_in[18];

  float* out_proj = (float*)d_out;                           // [B,S,D]
  float* sc = (float*)d_out + (size_t)kB * kS * kD;          // [B,H,S,S] scores

  char* wsp = (char*)d_ws;
  auto take = [&](size_t bytes) -> char* {
    char* p = wsp;
    wsp += (bytes + 255) & ~(size_t)255;
    return p;
  };
  const size_t nXD = (size_t)kRows * kD;      // 4M elements
  const size_t nW  = (size_t)kD * kD;         // 1M elements
  unsigned short* xq  = (unsigned short*)take(nXD * 2);
  unsigned short* xk  = (unsigned short*)take(nXD * 2);
  unsigned short* xv  = (unsigned short*)take(nXD * 2);
  unsigned short* wqb = (unsigned short*)take(nW * 2);
  unsigned short* wkb = (unsigned short*)take(nW * 2);
  unsigned short* wvb = (unsigned short*)take(nW * 2);
  unsigned short* wob = (unsigned short*)take(nW * 2);
  unsigned short* qh  = (unsigned short*)take(nXD * 2);      // [b][h][s][dk]
  unsigned short* kh  = (unsigned short*)take(nXD * 2);      // [b][h][s][dk]
  unsigned short* vt  = (unsigned short*)take(nXD * 2);      // [b][h][dk][s]
  unsigned short* xa  = (unsigned short*)take(nXD * 2);      // attn output bf16
  float*          Wc  = (float*)take(16 * 32 * sizeof(float));
  float*          bc  = (float*)take(16 * sizeof(float));
  unsigned short* pat = (unsigned short*)take((size_t)kB * kH * kSSq * 2);

  // 1) casts to bf16
  k_cast<<<(int)(nXD / 256), 256, 0, stream>>>(query,  xq, (int)nXD);
  k_cast<<<(int)(nXD / 256), 256, 0, stream>>>(key_in, xk, (int)nXD);
  k_cast<<<(int)(nXD / 256), 256, 0, stream>>>(value,  xv, (int)nXD);
  k_cast<<<(int)(nW / 256), 256, 0, stream>>>(Wq, wqb, (int)nW);
  k_cast<<<(int)(nW / 256), 256, 0, stream>>>(Wk, wkb, (int)nW);
  k_cast<<<(int)(nW / 256), 256, 0, stream>>>(Wv, wvb, (int)nW);
  k_cast<<<(int)(nW / 256), 256, 0, stream>>>(Wo, wob, (int)nW);

  // 2) Q/K/V projections (WMMA), head-split outputs; V transposed
  dim3 gG(kD / 128, kRows / 128);   // (8, 32)
  k_gemm<<<gG, 256, 0, stream>>>(xq, wqb, bq, qh, nullptr, kRows, kD, kD, 0);
  k_gemm<<<gG, 256, 0, stream>>>(xk, wkb, bk, kh, nullptr, kRows, kD, kD, 0);
  k_gemm<<<gG, 256, 0, stream>>>(xv, wvb, bv, vt, nullptr, kRows, kD, kD, 1);

  // 3) masked scores (WMMA) straight into d_out's scores region
  k_scores<<<dim3(kS / 128, kS / 128, kB * kH), 256, 0, stream>>>(qh, kh, mask, sc);

  // 4) fold 3 linear 1x1 convs into one [16x32] weight, apply in place
  k_foldconv<<<1, 512, 0, stream>>>(Cw1, Cb1, Cw2, Cb2, Cw3, Cb3, Wc, bc);
  k_conv<<<4096, 256, 0, stream>>>(sc, matx, Wc, bc);

  // 5) softmax rows -> bf16 p_attn
  k_softmax<<<(kB * kH * kS) / 8, 256, 0, stream>>>(sc, pat);

  // 6) x = P·V (WMMA)
  k_attnv<<<dim3(kS / 128, kB * kH), 256, 0, stream>>>(pat, vt, xa);

  // 7) output projection (WMMA) -> fp32 d_out
  k_gemm<<<gG, 256, 0, stream>>>(xa, wob, bo, nullptr, out_proj, kRows, kD, kD, 2);
}
